// GaugeNet_38680475468526
// MI455X (gfx1250) — matrix operand order repaired and checked by reading, not compiled, not run on hardware
//
#include <hip/hip_runtime.h>
#include <hip/hip_bf16.h>
#include <math.h>

typedef _Float16 half_t;
typedef __attribute__((ext_vector_type(16))) _Float16 v16h;
typedef __attribute__((ext_vector_type(8)))  float    v8f;
typedef __attribute__((ext_vector_type(4)))  unsigned int u4;
typedef __attribute__((ext_vector_type(4)))  unsigned int u32x4;
typedef __attribute__((ext_vector_type(8)))  int          i32x8;
typedef __attribute__((ext_vector_type(4)))  int          i32x4;

#if defined(__has_builtin)
#if __has_builtin(__builtin_amdgcn_tensor_load_to_lds)
#define USE_TDM 1
#endif
#endif
#ifndef USE_TDM
#define USE_TDM 0
#endif

union FragU { v16h h; u4 q[2]; };

// A-fragment (16x32 f16): 16 contiguous halves per lane
__device__ __forceinline__ v16h ldA(const half_t* p) {
  FragU f;
  f.q[0] = *(const u4*)(p);
  f.q[1] = *(const u4*)(p + 8);
  return f.h;
}
// B-fragment (32x16 f16): 16 contiguous halves per lane
__device__ __forceinline__ v16h ldB(const half_t* p) {
  FragU f;
  f.q[0] = *(const u4*)(p);
  f.q[1] = *(const u4*)(p + 8);
  return f.h;
}

// ---------------------------------------------------------------------------
// B-chunk staging: 32(K) x 256(cols) f16 from transposed weights into LDS,
// column stride 40 halves (32 data + 8 pad).  TDM: one 2D descriptor,
// 256 rows x 64B, pad 4 dwords per 16 dwords => 80B LDS stride.
// ---------------------------------------------------------------------------
#if USE_TDM
__device__ __forceinline__ void tdm_load_chunk(const half_t* gsrc,
                                               unsigned lds_byte_addr,
                                               int kstride) {
  unsigned long long ga = (unsigned long long)(uintptr_t)gsrc;
  u32x4 g0;
  g0[0] = 1u;                                    // count=1 (valid), user mode
  g0[1] = lds_byte_addr;                         // lds_addr
  g0[2] = (unsigned)(ga & 0xFFFFFFFFu);          // global_addr[31:0]
  g0[3] = (unsigned)((ga >> 32) & 0x1FFFFFFu)    // global_addr[56:32]
          | (2u << 30);                          // type=2 (image)
  i32x8 g1;
  g1[0] = (1 << 16)      // data_size = 2 bytes
        | (1 << 20)      // pad_enable
        | (3 << 22)      // pad_interval: 16 dwords (64B)
        | (3 << 25);     // pad_amount: 4 dwords (16B)
  g1[1] = (32 & 0xFFFF) << 16;   // tensor_dim0 = 32
  g1[2] = (256 << 16);           // tensor_dim1 = 256 (low 16)
  g1[3] = (32 << 16);            // tile_dim0 = 32
  g1[4] = 256;                   // tile_dim1 = 256, tile_dim2 = 0
  g1[5] = kstride;               // tensor_dim0_stride[31:0]
  g1[6] = 0;
  g1[7] = 0;
  i32x4 z4 = {0, 0, 0, 0};
#if defined(__clang_major__) && (__clang_major__ >= 23)
  i32x8 z8 = {0, 0, 0, 0, 0, 0, 0, 0};
  __builtin_amdgcn_tensor_load_to_lds(g0, g1, z4, z4, z8, 0);
#else
  __builtin_amdgcn_tensor_load_to_lds(g0, g1, z4, z4, 0);
#endif
}
#endif

__device__ __forceinline__ void stage_issue(const half_t* gsrc_chunk,
                                            half_t* lds_dst, int kstride,
                                            int tid) {
#if USE_TDM
  if (tid < 32) tdm_load_chunk(gsrc_chunk, (unsigned)(uintptr_t)lds_dst, kstride);
#else
  const half_t* s = gsrc_chunk + (size_t)tid * kstride;
  half_t* d = lds_dst + tid * 40;
  u4 a0 = *(const u4*)(s);
  u4 a1 = *(const u4*)(s + 8);
  u4 a2 = *(const u4*)(s + 16);
  u4 a3 = *(const u4*)(s + 24);
  *(u4*)(d) = a0; *(u4*)(d + 8) = a1; *(u4*)(d + 16) = a2; *(u4*)(d + 24) = a3;
#endif
}

__device__ __forceinline__ void stage_wait(int tid) {
#if USE_TDM
  if (tid < 32) __builtin_amdgcn_s_wait_tensorcnt(0);
#endif
  __syncthreads();
}

// ---------------------------------------------------------------------------
// Weight convert+transpose: src f32 (K x N, row-major) -> dst f16 (N x K)
// ---------------------------------------------------------------------------
__global__ void cvt_w(const float* __restrict__ src, half_t* __restrict__ dst,
                      int K, int N) {
  int t = blockIdx.x * 256 + threadIdx.x;
  if (t >= K * N) return;
  int n = t / K;
  int k = t - n * K;
  dst[t] = (half_t)src[k * N + n];
}

// ---------------------------------------------------------------------------
// Gauge features: h16[b, d*1024+i] = sum_{m,n} x[b,i,m]*H[m,n]*x[b,nbr[d,i],n]
// ---------------------------------------------------------------------------
__global__ __launch_bounds__(256) void gauge_k(const float* __restrict__ x,
                                               const float* __restrict__ Hm,
                                               const int*   __restrict__ nbr,
                                               half_t* __restrict__ h16) {
  __shared__ float xs[2048];
  const int b = blockIdx.x;
  const float* xb = x + (size_t)b * 2048;
  for (int j = threadIdx.x; j < 2048; j += 256) xs[j] = xb[j];
  __syncthreads();
  const float h00 = Hm[0], h01 = Hm[1], h10 = Hm[2], h11 = Hm[3];
  half_t* ob = h16 + (size_t)b * 4096;
  for (int f = threadIdx.x; f < 4096; f += 256) {
    int d = f >> 10, i = f & 1023;
    int j = nbr[(d << 10) + i];
    float xm0 = xs[2 * i], xm1 = xs[2 * i + 1];
    float xn0 = xs[2 * j], xn1 = xs[2 * j + 1];
    ob[f] = (half_t)(xm0 * (h00 * xn0 + h01 * xn1) +
                     xm1 * (h10 * xn0 + h11 * xn1));
  }
}

// ---------------------------------------------------------------------------
// Fused MLP. 8 waves/block = 4 row-tiles x 2 col-halves; each wave owns
// 16 batch rows x 128 output features (8 N-tiles, acc = 64 VGPRs).
// 1024 wave-tiles total for latency hiding.  B chunks staged via TDM
// (double-buffered, prefetch next during compute).  A re-reads hit L2
// (h16 = 64MB < 192MB L2).
// LDS: tiles 4*16*264*2 = 33792 | bstage 2*256*40*2 = 40960 | red 1024
// ---------------------------------------------------------------------------
#define OFF_TILE 0
#define OFF_B    33792
#define OFF_RED  74752
#define SMEM_BYTES 75776
#define BHALVES  10240   // halves per B buffer (256*40)

__global__ __launch_bounds__(256) void mlp_k(
    const half_t* __restrict__ h16,
    const half_t* __restrict__ wte, const float* __restrict__ be,
    const half_t* __restrict__ wt1, const float* __restrict__ bb1,
    const half_t* __restrict__ wt2, const float* __restrict__ bb2,
    const half_t* __restrict__ wt3, const float* __restrict__ bb3,
    const half_t* __restrict__ wt4, const float* __restrict__ bb4,
    const half_t* __restrict__ wt5, const float* __restrict__ bb5,
    const float*  __restrict__ wout, const float* __restrict__ bout,
    float* __restrict__ out) {
  extern __shared__ char smem_raw[];
  half_t* tile = (half_t*)(smem_raw + OFF_TILE);
  half_t* bst  = (half_t*)(smem_raw + OFF_B);
  float*  red  = (float*)(smem_raw + OFF_RED);

  const int tid  = threadIdx.x;
  const int lane = tid & 31;
  const int wave = tid >> 5;
  const int ch   = wave & 1;    // column half: features [ch*128, ch*128+128)
  const int rt   = wave >> 1;   // row tile 0..3
  const int mr   = lane & 15;
  const int ako  = (lane & 16) ? 8 : 0;    // C M-offset (accumulator rows)
  const int bko  = (lane & 16) ? 16 : 0;   // B K-offset
  const int row0 = blockIdx.x * 64 + rt * 16;
  half_t* mytile = tile + rt * (16 * 264);

  v8f acc[8];

  // ---------------- embedding: K = 4096, A from global h16 ----------------
  #pragma unroll
  for (int nt = 0; nt < 8; ++nt) {
    float bv = be[((ch * 8 + nt) << 4) + mr];
    #pragma unroll
    for (int r = 0; r < 8; ++r) acc[nt][r] = bv;
  }
  const half_t* arow = h16 + (size_t)(row0 + mr) * 4096;

  stage_issue(wte, bst, 4096, tid);            // chunk 0 -> buf 0
  stage_wait(tid);
  v16h a = ldA(arow + bko);

  for (int kc = 0; kc < 128; ++kc) {
    const int cur = kc & 1;
    if (kc + 1 < 128)
      stage_issue(wte + (kc + 1) * 32, bst + (cur ^ 1) * BHALVES, 4096, tid);
    v16h anext;
    if (kc + 1 < 128) anext = ldA(arow + (kc + 1) * 32 + bko);

    const half_t* bb = bst + cur * BHALVES;
    v16h bf[8];
    #pragma unroll
    for (int j = 0; j < 8; ++j)
      bf[j] = ldB(bb + (((ch * 8 + j) * 16) + mr) * 40 + bko);
    #pragma unroll
    for (int j = 0; j < 8; ++j)
      acc[j] = __builtin_amdgcn_wmma_f32_16x16x32_f16(
          false, a, false, bf[j], (short)0, acc[j], false, false);
    a = anext;
    if (kc + 1 < 128) stage_wait(tid);
  }
  // store f16 tile half (no activation after embedding); next layer's
  // stage_wait barrier orders these writes vs partner-wave reads.
  #pragma unroll
  for (int nt = 0; nt < 8; ++nt)
    #pragma unroll
    for (int r = 0; r < 8; ++r)
      mytile[(ako + r) * 264 + ((ch * 8 + nt) << 4) + mr] = (half_t)acc[nt][r];

  // ---------------- hidden layers: K = 256, A from shared LDS tile ---------
  auto hidden = [&](const half_t* __restrict__ Wt,
                    const float* __restrict__ bias, bool act) {
    #pragma unroll
    for (int nt = 0; nt < 8; ++nt) {
      float bv = bias[((ch * 8 + nt) << 4) + mr];
      #pragma unroll
      for (int r = 0; r < 8; ++r) acc[nt][r] = bv;
    }
    stage_issue(Wt, bst, 256, tid);
    stage_wait(tid);           // also orders previous tile writes vs reads
    for (int kc = 0; kc < 8; ++kc) {
      const int cur = kc & 1;
      if (kc + 1 < 8)
        stage_issue(Wt + (kc + 1) * 32, bst + (cur ^ 1) * BHALVES, 256, tid);
      v16h av = ldA(mytile + mr * 264 + kc * 32 + bko);
      const half_t* bb = bst + cur * BHALVES;
      v16h bf[8];
      #pragma unroll
      for (int j = 0; j < 8; ++j)
        bf[j] = ldB(bb + (((ch * 8 + j) * 16) + mr) * 40 + bko);
      #pragma unroll
      for (int j = 0; j < 8; ++j)
        acc[j] = __builtin_amdgcn_wmma_f32_16x16x32_f16(
            false, av, false, bf[j], (short)0, acc[j], false, false);
      if (kc + 1 < 8) stage_wait(tid);
    }
    // tile is shared by the wave pair: wait until both finished reading
    // the previous layer's values before overwriting.
    __syncthreads();
    #pragma unroll
    for (int nt = 0; nt < 8; ++nt) {
      #pragma unroll
      for (int r = 0; r < 8; ++r) {
        float v = acc[nt][r];
        if (act) v = v / (1.0f + __expf(-v));   // silu
        mytile[(ako + r) * 264 + ((ch * 8 + nt) << 4) + mr] = (half_t)v;
      }
    }
  };

  hidden(wt1, bb1, true);    // silu(h @ W_pre1 + b)
  hidden(wt2, bb2, false);   //       @ W_pre2 + b
  hidden(wt3, bb3, true);    // silu(  @ W_net1 + b)
  hidden(wt4, bb4, true);    // silu(  @ W_net2 + b)
  hidden(wt5, bb5, true);    // silu(  @ W_post1 + b)

  // ---------------- head: 256 -> 1 ----------------
  __syncthreads();   // partner tile writes visible
  // wave pair = 64 lanes per row-tile: row = p&15, k-segment = p>>4 (4x64)
  const int p    = ch * 32 + lane;
  const int hrow = p & 15;
  const int kseg = p >> 4;
  float s = 0.0f;
  for (int k = 0; k < 64; ++k)
    s += (float)mytile[hrow * 264 + kseg * 64 + k] * wout[kseg * 64 + k];
  red[rt * 64 + p] = s;
  __syncthreads();
  if (ch == 0 && lane < 16) {
    float tot = red[rt * 64 + mr] + red[rt * 64 + 16 + mr] +
                red[rt * 64 + 32 + mr] + red[rt * 64 + 48 + mr];
    out[row0 + mr] = tot + bout[0];
  }
}

// ---------------------------------------------------------------------------
extern "C" void kernel_launch(void* const* d_in, const int* in_sizes, int n_in,
                              void* d_out, int out_size, void* d_ws, size_t ws_size,
                              hipStream_t stream) {
  (void)in_sizes; (void)n_in; (void)out_size; (void)ws_size;
  const float* x    = (const float*)d_in[0];
  const float* Hm   = (const float*)d_in[1];
  const int*   nbr  = (const int*)  d_in[2];
  const float* W_emb = (const float*)d_in[3];  const float* b_emb = (const float*)d_in[4];
  const float* W1 = (const float*)d_in[5];     const float* b1 = (const float*)d_in[6];
  const float* W2 = (const float*)d_in[7];     const float* b2 = (const float*)d_in[8];
  const float* W3 = (const float*)d_in[9];     const float* b3 = (const float*)d_in[10];
  const float* W4 = (const float*)d_in[11];    const float* b4 = (const float*)d_in[12];
  const float* W5 = (const float*)d_in[13];    const float* b5 = (const float*)d_in[14];
  const float* Wout = (const float*)d_in[15];  const float* bout = (const float*)d_in[16];
  float* out = (float*)d_out;

  // workspace layout (f16 elements)
  half_t* h16 = (half_t*)d_ws;                       // 8192*4096
  half_t* wte = h16 + (size_t)8192 * 4096;           // 4096*256
  half_t* wt1 = wte + (size_t)4096 * 256;            // 256*256 each
  half_t* wt2 = wt1 + 256 * 256;
  half_t* wt3 = wt2 + 256 * 256;
  half_t* wt4 = wt3 + 256 * 256;
  half_t* wt5 = wt4 + 256 * 256;

  cvt_w<<<(4096 * 256 + 255) / 256, 256, 0, stream>>>(W_emb, wte, 4096, 256);
  cvt_w<<<(256 * 256 + 255) / 256, 256, 0, stream>>>(W1, wt1, 256, 256);
  cvt_w<<<(256 * 256 + 255) / 256, 256, 0, stream>>>(W2, wt2, 256, 256);
  cvt_w<<<(256 * 256 + 255) / 256, 256, 0, stream>>>(W3, wt3, 256, 256);
  cvt_w<<<(256 * 256 + 255) / 256, 256, 0, stream>>>(W4, wt4, 256, 256);
  cvt_w<<<(256 * 256 + 255) / 256, 256, 0, stream>>>(W5, wt5, 256, 256);

  gauge_k<<<8192, 256, 0, stream>>>(x, Hm, nbr, h16);

  (void)hipFuncSetAttribute((const void*)mlp_k,
                            hipFuncAttributeMaxDynamicSharedMemorySize,
                            SMEM_BYTES);
  mlp_k<<<128, 256, SMEM_BYTES, stream>>>(h16, wte, b_emb, wt1, b1, wt2, b2,
                                          wt3, b3, wt4, b4, wt5, b5,
                                          Wout, bout, out);
}
